// SparseAttention_63230508531950
// MI455X (gfx1250) — compile-verified, matrix operand order checked by reading, not run on hardware
//
#include <hip/hip_runtime.h>
#include <math.h>

// ---------------- CDNA5 WMMA types ----------------
typedef __attribute__((ext_vector_type(16))) __bf16 v16bf;
typedef __attribute__((ext_vector_type(8)))  __bf16 v8bf;
typedef __attribute__((ext_vector_type(8)))  float  v8f;

#define BB      2
#define TT      2048
#define CC      1024
#define HH      16
#define HD      64
#define NIDX    (64 * TT)          // 131072 index pairs per (b,h)
#define BH      (BB * HH)          // 32
#define MROWS   (BB * TT)          // 4096
#define NQKV    (3 * CC)           // 3072
#define MRW     (TT / 32)          // mask row words = 64

#define WMMA_BF16(A, B, C) \
    __builtin_amdgcn_wmma_f32_16x16x32_bf16(false, (A), false, (B), (short)0, (C), false, false)

static __device__ __forceinline__ v16bf frag_load(const __bf16* base, int lane) {
    // Row-major fragment load matching CDNA5 wave32 16x32 bf16 A/B layout:
    // element i -> K = (i<8 ? i : 8+i) + (lane>=16 ? 8 : 0)
    const int off = (lane >= 16) ? 8 : 0;
    v8bf lo = *(const v8bf*)(base + off);
    v8bf hi = *(const v8bf*)(base + 16 + off);
    v16bf r;
#pragma unroll
    for (int i = 0; i < 8; ++i) { r[i] = lo[i]; r[i + 8] = hi[i]; }
    return r;
}

static __device__ __forceinline__ float half16_max(float x) {
#pragma unroll
    for (int m = 1; m < 16; m <<= 1) x = fmaxf(x, __shfl_xor(x, m, 32));
    return x;
}
static __device__ __forceinline__ float half16_sum(float x) {
#pragma unroll
    for (int m = 1; m < 16; m <<= 1) x += __shfl_xor(x, m, 32);
    return x;
}

// ---------------- f32 -> bf16 conversion ----------------
__global__ void sattn_cvt_bf16(const float* __restrict__ src, __bf16* __restrict__ dst, int n) {
    int i = blockIdx.x * blockDim.x + threadIdx.x;
    int stride = gridDim.x * blockDim.x;
    for (; i < n; i += stride) dst[i] = (__bf16)src[i];
}

// ---------------- sparse mask build ----------------
__global__ void sattn_mask_zero(unsigned* __restrict__ m, int nwords) {
    int i = blockIdx.x * blockDim.x + threadIdx.x;
    int stride = gridDim.x * blockDim.x;
    for (; i < nwords; i += stride) m[i] = 0u;
}
__global__ void sattn_mask_scatter(const int* __restrict__ idx, unsigned* __restrict__ m) {
    int i = blockIdx.x * blockDim.x + threadIdx.x;      // over BH * NIDX entries
    if (i >= BH * NIDX) return;
    int bh = i / NIDX;
    int qi = idx[(size_t)i * 2 + 0];
    int ki = idx[(size_t)i * 2 + 1];
    atomicOr(&m[((size_t)bh * TT + qi) * MRW + (ki >> 5)], 1u << (ki & 31));
}

// ---------------- QKV projection GEMM (M=4096, N=3072, K=1024) ----------------
// One wave -> 32x64 output tile: 8 independent WMMAs per k-step, register
// double-buffered fragments (ping-pong) so loads for step n+1 overlap WMMAs of step n.
__global__ __launch_bounds__(256) void sattn_qkv_gemm(
    const __bf16* __restrict__ xbf, const __bf16* __restrict__ wbf,
    const float* __restrict__ bqkv,
    __bf16* __restrict__ qbuf, __bf16* __restrict__ kbuf, __bf16* __restrict__ vbuf)
{
    const int wave = threadIdx.x >> 5, lane = threadIdx.x & 31;
    const int nl = lane & 15;
    const int tile = blockIdx.x * 8 + wave;
    const int NTN = NQKV / 64;                 // 48
    const int m0 = (tile / NTN) * 32;
    const int n0 = (tile % NTN) * 64;

    v8f acc[2][4];
#pragma unroll
    for (int i = 0; i < 2; ++i)
#pragma unroll
        for (int j = 0; j < 4; ++j) acc[i][j] = {};

    const __bf16* ap0 = xbf + (size_t)(m0 + nl) * CC;
    const __bf16* ap1 = xbf + (size_t)(m0 + 16 + nl) * CC;
    const __bf16* bp[4];
#pragma unroll
    for (int j = 0; j < 4; ++j) bp[j] = wbf + (size_t)(n0 + j * 16 + nl) * CC;

    v16bf aX[2], bX[4], aY[2], bY[4];
    aX[0] = frag_load(ap0, lane);
    aX[1] = frag_load(ap1, lane);
#pragma unroll
    for (int j = 0; j < 4; ++j) bX[j] = frag_load(bp[j], lane);

    auto gstep = [&](int k0, v16bf (&aC)[2], v16bf (&bC)[4],
                     v16bf (&aN)[2], v16bf (&bN)[4]) {
        int kn = k0 + 32; if (kn >= CC) kn = 0;   // harmless wrap on final prefetch
        // issue next-step loads first; they complete under this step's WMMAs
        aN[0] = frag_load(ap0 + kn, lane);
        aN[1] = frag_load(ap1 + kn, lane);
#pragma unroll
        for (int j = 0; j < 4; ++j) bN[j] = frag_load(bp[j] + kn, lane);
        __builtin_prefetch(ap0 + kn + 32, 0, 1);   // stream A two steps ahead
        __builtin_prefetch(ap1 + kn + 32, 0, 1);
#pragma unroll
        for (int i = 0; i < 2; ++i)
#pragma unroll
            for (int j = 0; j < 4; ++j)
                acc[i][j] = WMMA_BF16(aC[i], bC[j], acc[i][j]);
    };
    for (int k0 = 0; k0 < CC; k0 += 64) {
        gstep(k0,      aX, bX, aY, bY);
        gstep(k0 + 32, aY, bY, aX, bX);
    }

    const int mhalf = (lane >= 16) ? 8 : 0;
#pragma unroll
    for (int j = 0; j < 4; ++j) {
        const int n = n0 + j * 16 + nl;
        const int h = n / 192, r = n - h * 192;
        const float bias = bqkv[n];
#pragma unroll
        for (int i = 0; i < 2; ++i) {
#pragma unroll
            for (int v = 0; v < 8; ++v) {
                const int mrow = m0 + i * 16 + v + mhalf;
                const int b = mrow >> 11, t = mrow & (TT - 1);
                const size_t bh = (size_t)b * HH + h;
                const float val = acc[i][j][v] + bias;
                if (r < HD) {
                    qbuf[(bh * TT + t) * HD + r] = (__bf16)(val * 0.125f); // fold HD^-0.5
                } else if (r < 2 * HD) {
                    kbuf[(bh * TT + t) * HD + (r - HD)] = (__bf16)val;
                } else {
                    vbuf[(bh * HD + (r - 2 * HD)) * TT + t] = (__bf16)val; // V transposed
                }
            }
        }
    }
}

// ---------------- fused masked-softmax flash attention ----------------
// One wave per (b,h, 16-row q tile); 32-key steps. V fragments + mask words are
// issued at the top of each step, next-step K fragments right after the score
// WMMAs, so the VALU-heavy online softmax hides all global latency.
__global__ __launch_bounds__(256) void sattn_attention(
    const __bf16* __restrict__ qbuf, const __bf16* __restrict__ kbuf,
    const __bf16* __restrict__ vbuf, const unsigned* __restrict__ maskbuf,
    __bf16* __restrict__ abuf)
{
    __shared__ __align__(16) __bf16 plds[8][16][40];   // 16x32 P tile, padded rows

    const int wave = threadIdx.x >> 5, lane = threadIdx.x & 31;
    const int nl = lane & 15;
    const int mhalf = (lane >= 16) ? 8 : 0;
    const int id = blockIdx.x * 8 + wave;
    const int bh = id >> 7;                  // id / (T/16)
    const int q0 = (id & 127) * 16;

    const __bf16* qrow = qbuf + ((size_t)bh * TT + q0 + nl) * HD;
    const v16bf qA0 = frag_load(qrow, lane);        // d = 0..31
    const v16bf qA1 = frag_load(qrow + 32, lane);   // d = 32..63

    v8f O[4];
#pragma unroll
    for (int j = 0; j < 4; ++j) O[j] = {};
    float rmax[8], rsum[8];
#pragma unroll
    for (int v = 0; v < 8; ++v) { rmax[v] = -1e30f; rsum[v] = 0.f; }

    const unsigned* mrow = maskbuf + ((size_t)bh * TT + q0) * MRW;
    const __bf16* kbase = kbuf + (size_t)bh * TT * HD;
    const __bf16* vbase = vbuf + (size_t)bh * HD * TT;

    auto load_kfrags = [&](int k0, v16bf (&kf)[4]) {
        const __bf16* kp0 = kbase + (size_t)(k0 + nl) * HD;
        const __bf16* kp1 = kbase + (size_t)(k0 + 16 + nl) * HD;
        kf[0] = frag_load(kp0, lane);
        kf[1] = frag_load(kp0 + 32, lane);
        kf[2] = frag_load(kp1, lane);
        kf[3] = frag_load(kp1 + 32, lane);
    };

    v16bf kX[4], kY[4];
    load_kfrags(0, kX);

    auto step = [&](int k0, v16bf (&kC)[4], v16bf (&kN)[4]) {
        // ---- issue V fragments + mask words early (consumed after softmax) ----
        v16bf vF[4];
#pragma unroll
        for (int j = 0; j < 4; ++j)
            vF[j] = frag_load(vbase + (size_t)(j * 16 + nl) * TT + k0, lane);
        const int kw = k0 >> 5;
        unsigned wbits[8];
#pragma unroll
        for (int v = 0; v < 8; ++v) wbits[v] = mrow[(size_t)(v + mhalf) * MRW + kw];

        // ---- scores for keys [k0, k0+32) with resident K fragments ----
        v8f S0 = {}, S1 = {};
        S0 = WMMA_BF16(qA0, kC[0], S0);
        S0 = WMMA_BF16(qA1, kC[1], S0);
        S1 = WMMA_BF16(qA0, kC[2], S1);
        S1 = WMMA_BF16(qA1, kC[3], S1);

        // ---- prefetch next step's K fragments ----
        int kn = k0 + 32; if (kn >= TT) kn = 0;
        load_kfrags(kn, kN);

        // ---- bit-mask + online softmax (one u32 word covers this 32-key step) ----
#pragma unroll
        for (int v = 0; v < 8; ++v) {
            const unsigned w = wbits[v];
            const bool b0 = (w >> nl) & 1u;
            const bool b1 = (w >> (16 + nl)) & 1u;
            const float s0 = b0 ? S0[v] : -1e30f;
            const float s1 = b1 ? S1[v] : -1e30f;
            const float tmax = half16_max(fmaxf(s0, s1));
            const float nmax = fmaxf(rmax[v], tmax);
            const float corr = __expf(rmax[v] - nmax);
            rmax[v] = nmax;
            const float p0 = b0 ? __expf(s0 - nmax) : 0.f;
            const float p1 = b1 ? __expf(s1 - nmax) : 0.f;
            rsum[v] = rsum[v] * corr + half16_sum(p0 + p1);
#pragma unroll
            for (int j = 0; j < 4; ++j) O[j][v] *= corr;
            plds[wave][v + mhalf][nl]      = (__bf16)p0;
            plds[wave][v + mhalf][16 + nl] = (__bf16)p1;
        }

        // ---- C-layout -> A-layout transpose of P through LDS (same-wave DS in-order) ----
        v16bf pA;
        {
            v8bf lo = *(const v8bf*)&plds[wave][nl][mhalf];
            v8bf hi = *(const v8bf*)&plds[wave][nl][16 + mhalf];
#pragma unroll
            for (int i = 0; i < 8; ++i) { pA[i] = lo[i]; pA[i + 8] = hi[i]; }
        }

        // ---- O(16x64) += P(16x32) x V(32x64) ----
#pragma unroll
        for (int j = 0; j < 4; ++j) O[j] = WMMA_BF16(pA, vF[j], O[j]);
    };

    for (int k0 = 0; k0 < TT; k0 += 64) {
        step(k0,      kX, kY);
        step(k0 + 32, kY, kX);
    }

    // ---- epilogue: normalize, store to [b][t][h*64+d] bf16 ----
    const int b = bh >> 4, h = bh & 15;
#pragma unroll
    for (int v = 0; v < 8; ++v) {
        const int t = q0 + v + mhalf;
        const float inv = 1.f / fmaxf(rsum[v], 1e-20f);
#pragma unroll
        for (int j = 0; j < 4; ++j) {
            abuf[((size_t)(b * TT + t)) * CC + h * HD + j * 16 + nl] = (__bf16)(O[j][v] * inv);
        }
    }
}

// ---------------- output projection GEMM (M=4096, N=1024, K=1024), f32 out ----------------
__global__ __launch_bounds__(256) void sattn_out_gemm(
    const __bf16* __restrict__ abuf, const __bf16* __restrict__ wbf,
    const float* __restrict__ bout, float* __restrict__ out)
{
    const int wave = threadIdx.x >> 5, lane = threadIdx.x & 31;
    const int nl = lane & 15;
    const int tile = blockIdx.x * 8 + wave;
    const int NTN = CC / 64;                   // 16
    const int m0 = (tile / NTN) * 32;
    const int n0 = (tile % NTN) * 64;

    v8f acc[2][4];
#pragma unroll
    for (int i = 0; i < 2; ++i)
#pragma unroll
        for (int j = 0; j < 4; ++j) acc[i][j] = {};

    const __bf16* ap0 = abuf + (size_t)(m0 + nl) * CC;
    const __bf16* ap1 = abuf + (size_t)(m0 + 16 + nl) * CC;
    const __bf16* bp[4];
#pragma unroll
    for (int j = 0; j < 4; ++j) bp[j] = wbf + (size_t)(n0 + j * 16 + nl) * CC;

    v16bf aX[2], bX[4], aY[2], bY[4];
    aX[0] = frag_load(ap0, lane);
    aX[1] = frag_load(ap1, lane);
#pragma unroll
    for (int j = 0; j < 4; ++j) bX[j] = frag_load(bp[j], lane);

    auto gstep = [&](int k0, v16bf (&aC)[2], v16bf (&bC)[4],
                     v16bf (&aN)[2], v16bf (&bN)[4]) {
        int kn = k0 + 32; if (kn >= CC) kn = 0;
        aN[0] = frag_load(ap0 + kn, lane);
        aN[1] = frag_load(ap1 + kn, lane);
#pragma unroll
        for (int j = 0; j < 4; ++j) bN[j] = frag_load(bp[j] + kn, lane);
        __builtin_prefetch(ap0 + kn + 32, 0, 1);
        __builtin_prefetch(ap1 + kn + 32, 0, 1);
#pragma unroll
        for (int i = 0; i < 2; ++i)
#pragma unroll
            for (int j = 0; j < 4; ++j)
                acc[i][j] = WMMA_BF16(aC[i], bC[j], acc[i][j]);
    };
    for (int k0 = 0; k0 < CC; k0 += 64) {
        gstep(k0,      aX, bX, aY, bY);
        gstep(k0 + 32, aY, bY, aX, bX);
    }

    const int mhalf = (lane >= 16) ? 8 : 0;
#pragma unroll
    for (int j = 0; j < 4; ++j) {
        const int n = n0 + j * 16 + nl;
        const float bias = bout[n];
#pragma unroll
        for (int i = 0; i < 2; ++i) {
#pragma unroll
            for (int v = 0; v < 8; ++v) {
                const int mrow = m0 + i * 16 + v + mhalf;
                out[(size_t)mrow * CC + n] = acc[i][j][v] + bias;
            }
        }
    }
}

// ---------------- host-side launch ----------------
extern "C" void kernel_launch(void* const* d_in, const int* in_sizes, int n_in,
                              void* d_out, int out_size, void* d_ws, size_t ws_size,
                              hipStream_t stream) {
    (void)in_sizes; (void)n_in; (void)out_size; (void)ws_size;
    const float* x    = (const float*)d_in[0];
    const int*   idx  = (const int*)d_in[1];
    const float* Wqkv = (const float*)d_in[2];
    const float* bqkv = (const float*)d_in[3];
    const float* Wout = (const float*)d_in[4];
    const float* bout = (const float*)d_in[5];
    float* out = (float*)d_out;

    char* ws = (char*)d_ws;
    size_t off = 0;
    auto take = [&](size_t bytes) { char* p = ws + off; off += (bytes + 255) & ~(size_t)255; return p; };
    __bf16* xbf    = (__bf16*)take((size_t)MROWS * CC * 2);
    __bf16* wqkvbf = (__bf16*)take((size_t)NQKV * CC * 2);
    __bf16* woutbf = (__bf16*)take((size_t)CC * CC * 2);
    __bf16* qbuf   = (__bf16*)take((size_t)BH * TT * HD * 2);
    __bf16* kbuf   = (__bf16*)take((size_t)BH * TT * HD * 2);
    __bf16* vbuf   = (__bf16*)take((size_t)BH * TT * HD * 2);
    __bf16* abuf   = (__bf16*)take((size_t)MROWS * CC * 2);
    unsigned* mask = (unsigned*)take((size_t)BH * TT * MRW * 4);

    sattn_cvt_bf16<<<4096, 256, 0, stream>>>(x, xbf, MROWS * CC);
    sattn_cvt_bf16<<<2048, 256, 0, stream>>>(Wqkv, wqkvbf, NQKV * CC);
    sattn_cvt_bf16<<<1024, 256, 0, stream>>>(Wout, woutbf, CC * CC);

    sattn_mask_zero<<<4096, 256, 0, stream>>>(mask, BH * TT * MRW);
    sattn_mask_scatter<<<(BH * NIDX + 255) / 256, 256, 0, stream>>>(idx, mask);

    // (4096/32) m-tiles * 48 n-tiles = 6144 waves, 8 waves/block
    sattn_qkv_gemm<<<768, 256, 0, stream>>>(xbf, wqkvbf, bqkv, qbuf, kbuf, vbuf);

    // BH * (T/16) = 4096 waves
    sattn_attention<<<512, 256, 0, stream>>>(qbuf, kbuf, vbuf, mask, abuf);

    // (4096/32) * 16 = 2048 waves
    sattn_out_gemm<<<256, 256, 0, stream>>>(abuf, woutbf, bout, out);
}
